// GQAttention_1082331758935
// MI455X (gfx1250) — compile-verified, matrix operand order checked by reading, not run on hardware
//
#include <hip/hip_runtime.h>
#include <cstdint>

// ---------------------------------------------------------------------------
// Types
// ---------------------------------------------------------------------------
typedef unsigned short u16;
typedef __attribute__((ext_vector_type(16))) __bf16 bf16x16;
typedef __attribute__((ext_vector_type(8)))  float  v8f;
typedef __attribute__((ext_vector_type(8)))  u16    u16x8;
typedef __attribute__((ext_vector_type(16))) u16    u16x16;
typedef __attribute__((ext_vector_type(4)))  unsigned int u32x4;
typedef __attribute__((ext_vector_type(8)))  int    i32x8;
typedef __attribute__((ext_vector_type(4)))  int    i32x4;

#define L_SEQ    4096
#define IDIM     2048
#define ODIM     2048
#define N_HEADS  8
#define N_GROUPS 2
#define HEAD_DIM 256
#define ROPE_DIM 64
#define N_KV     512

// ---------------------------------------------------------------------------
// Helpers
// ---------------------------------------------------------------------------
__device__ __forceinline__ u16 f2bf(float f) {
  union { float f; unsigned u; } c; c.f = f;
  unsigned u = c.u;
  unsigned r = u + 0x7FFFu + ((u >> 16) & 1u);   // round-to-nearest-even
  return (u16)(r >> 16);
}

// Load a 16x32 bf16 A-fragment (or Bt-fragment) from a row-major [16 x ld]
// tile. ISA 16-bit A layout: lanes 0-15: M=lane, K 0-7 then 16-23;
// lanes 16-31: M=lane-16, K 8-15 then 24-31.
__device__ __forceinline__ bf16x16 load_frag(const u16* base, int ld) {
  const int lane = threadIdx.x & 31;
  const int row  = lane & 15;
  const int kb   = (lane >> 4) << 3;              // 0 or 8
  const u16* p = base + (size_t)row * ld + kb;
  union { u16x16 w; u16x8 h[2]; } u;
  u.h[0] = *(const u16x8*)(p);                    // K kb..kb+7
  u.h[1] = *(const u16x8*)(p + 16);               // K kb+16..kb+23
  return __builtin_bit_cast(bf16x16, u.w);
}

__device__ __forceinline__ v8f wmma_bf16(bf16x16 a, bf16x16 b, v8f c) {
  return __builtin_amdgcn_wmma_f32_16x16x32_bf16(false, a, false, b,
                                                 (short)0, c, false, false);
}

// C/D 16x16 f32 layout: VGPR r -> M = r + 8*(lane>=16); N = lane&15
__device__ __forceinline__ void store_tile(float* C, int ldc, int m0, int n0, v8f acc) {
  const int lane = threadIdx.x & 31;
  const int col = n0 + (lane & 15);
  const int rb  = m0 + ((lane >> 4) << 3);
#pragma unroll
  for (int r = 0; r < 8; ++r) C[(size_t)(rb + r) * ldc + col] = acc[r];
}

// ---------------------------------------------------------------------------
// Tensor Data Mover: async 2-D bf16 tile load, global -> LDS.
// D# per CDNA5 ISA ch.8: group0 = {count, lds_addr, global_addr, type=2},
// group1 = {data_size, tensor dims, tile dims, dim0 stride}, groups 2/3 = 0.
// TDM packs tile rows contiguously in LDS (tile_d0 elements per row).
// 6-arg builtin variant (clang-23 / therock headers).
// ---------------------------------------------------------------------------
__device__ __forceinline__ void tdm_load_2d(unsigned lds_addr, const u16* gptr,
                                            unsigned tile_d0, unsigned tile_d1,
                                            unsigned tensor_d0, unsigned tensor_d1,
                                            unsigned stride0_elems) {
  unsigned long long ga = (unsigned long long)(uintptr_t)gptr;
  u32x4 g0;
  g0[0] = 1u;                                          // count=1, is_restore=0
  g0[1] = lds_addr;                                    // LDS byte address
  g0[2] = (unsigned)(ga & 0xFFFFFFFFu);                // global_addr[31:0]
  g0[3] = (unsigned)((ga >> 32) & 0x1FFFFFFu) | (2u << 30);  // addr[56:32] | type=2
  i32x8 g1;
  g1[0] = (int)(1u << 16);                             // data_size=1 (2 bytes)
  g1[1] = (int)((tensor_d0 & 0xFFFFu) << 16);          // tensor_dim0[15:0] @bit48
  g1[2] = (int)(((tensor_d0 >> 16) & 0xFFFFu) |        // tensor_dim0[31:16]
                ((tensor_d1 & 0xFFFFu) << 16));        // tensor_dim1[15:0] @bit80
  g1[3] = (int)(((tensor_d1 >> 16) & 0xFFFFu) |        // tensor_dim1[31:16]
                ((tile_d0 & 0xFFFFu) << 16));          // tile_dim0 @bit112
  g1[4] = (int)(tile_d1 & 0xFFFFu);                    // tile_dim1 (tile_dim2=0)
  g1[5] = (int)stride0_elems;                          // tensor_dim0_stride[31:0]
  g1[6] = 0;                                           // stride0[47:32], stride1 lo
  g1[7] = 0;                                           // stride1 hi
  i32x4 gz4 = {0, 0, 0, 0};
  i32x8 gz8 = {0, 0, 0, 0, 0, 0, 0, 0};
  __builtin_amdgcn_tensor_load_to_lds(g0, g1, gz4, gz4, gz8, 0);
}

// ---------------------------------------------------------------------------
// Elementwise converters
// ---------------------------------------------------------------------------
__global__ void __launch_bounds__(256) cvt_bf16_kernel(const float* __restrict__ X,
                                                       u16* __restrict__ Xb, int n) {
  int i = blockIdx.x * 256 + threadIdx.x;
  if (i < n) Xb[i] = f2bf(X[i]);
}

// W (K x N, fp32 row-major)  ->  Wt (N x K, bf16 row-major)
__global__ void __launch_bounds__(256) transpose_w_kernel(const float* __restrict__ W,
                                                          u16* __restrict__ Wt,
                                                          int K, int N) {
  int n = blockIdx.x * 32 + (threadIdx.x & 31);
  int k = blockIdx.y * 8  + (threadIdx.x >> 5);
  if (n < N && k < K) Wt[(size_t)n * K + k] = f2bf(W[(size_t)k * N + n]);
}

// vraw (L x 512 fp32) -> Vt (512 x L bf16)
__global__ void __launch_bounds__(256) vtrans_kernel(const float* __restrict__ vraw,
                                                     u16* __restrict__ Vt) {
  int i = blockIdx.x * 256 + threadIdx.x;           // over L*512
  if (i >= L_SEQ * N_KV) return;
  int l = i >> 9, c = i & 511;
  Vt[(size_t)c * L_SEQ + l] = f2bf(vraw[i]);
}

// ---------------------------------------------------------------------------
// bf16 WMMA GEMM with TDM double-buffered LDS staging.
//   C (MxN fp32) = A (MxK bf16 row-major) x Bt^T, Bt stored (N x K) row-major.
// Block: 256 threads = 8 waves (4 M x 2 N), wave tile 32x32, block tile 128x64.
// Per K-step each wave TDM-loads its 16-row A slice + 8-row B slice into LDS
// (rows packed contiguously, ld = 32), waits TENSORcnt, barriers, computes.
// ---------------------------------------------------------------------------
__global__ void __launch_bounds__(256) gemm_bf16_tdm_kernel(const u16* __restrict__ A,
                                                            const u16* __restrict__ Bt,
                                                            float* __restrict__ C,
                                                            int M, int N, int K) {
  __shared__ __align__(16) u16 ldsA[2][128 * 32];   // 16 KB
  __shared__ __align__(16) u16 ldsB[2][64 * 32];    //  8 KB
  const int wave = threadIdx.x >> 5;
  const int wm = wave & 3, wn = wave >> 2;
  const int mblk = blockIdx.x * 128;
  const int nblk = blockIdx.y * 64;

  const unsigned aBase = (unsigned)(uintptr_t)(&ldsA[0][0]);
  const unsigned bBase = (unsigned)(uintptr_t)(&ldsB[0][0]);

  auto tdm_issue = [&](int buf, int k0) {
    // A slice: rows [mblk + wave*16, +16), 32 K-cols -> contiguous 1 KB in LDS
    tdm_load_2d(aBase + (unsigned)(buf * 128 * 32 + wave * 16 * 32) * 2u,
                A + (size_t)(mblk + wave * 16) * K + k0,
                32u, 16u, (unsigned)K, (unsigned)M, (unsigned)K);
    // B slice: rows [nblk + wave*8, +8)
    tdm_load_2d(bBase + (unsigned)(buf * 64 * 32 + wave * 8 * 32) * 2u,
                Bt + (size_t)(nblk + wave * 8) * K + k0,
                32u, 8u, (unsigned)K, (unsigned)N, (unsigned)K);
  };

  v8f a00 = {}, a01 = {}, a10 = {}, a11 = {};

  tdm_issue(0, 0);
  __builtin_amdgcn_s_wait_tensorcnt(0);
  __syncthreads();

  const int nsteps = K >> 5;
  for (int s = 0; s < nsteps; ++s) {
    const int cur = s & 1;
    if (s + 1 < nsteps) tdm_issue(cur ^ 1, (s + 1) << 5);

    const u16* As = &ldsA[cur][0];
    const u16* Bs = &ldsB[cur][0];
    bf16x16 af0 = load_frag(As + (wm * 32) * 32, 32);
    bf16x16 af1 = load_frag(As + (wm * 32 + 16) * 32, 32);
    bf16x16 bf0 = load_frag(Bs + (wn * 32) * 32, 32);
    bf16x16 bf1 = load_frag(Bs + (wn * 32 + 16) * 32, 32);
    a00 = wmma_bf16(af0, bf0, a00);
    a01 = wmma_bf16(af0, bf1, a01);
    a10 = wmma_bf16(af1, bf0, a10);
    a11 = wmma_bf16(af1, bf1, a11);

    if (s + 1 < nsteps) __builtin_amdgcn_s_wait_tensorcnt(0);
    __syncthreads();   // slices visible WG-wide; protects buffer reuse
  }
  store_tile(C, N, mblk + wm * 32,      nblk + wn * 32,      a00);
  store_tile(C, N, mblk + wm * 32,      nblk + wn * 32 + 16, a01);
  store_tile(C, N, mblk + wm * 32 + 16, nblk + wn * 32,      a10);
  store_tile(C, N, mblk + wm * 32 + 16, nblk + wn * 32 + 16, a11);
}

// ---------------------------------------------------------------------------
// RMSNorm (over 256) + partial RoPE (first 64 dims), fp32 in -> bf16 out.
// One wave per (row l, head h). Lane holds d = lane + 32*i, i=0..7, so the
// RoPE pair (d, d+32) is (i=0, i=1) within the same lane.
// ---------------------------------------------------------------------------
__global__ void __launch_bounds__(256) norm_rope_kernel(const float* __restrict__ raw,
                                                        const float* __restrict__ gamma,
                                                        const float* __restrict__ cosT,
                                                        const float* __restrict__ sinT,
                                                        u16* __restrict__ out,
                                                        int H, int inStride, int inLd, int outLd) {
  const int wave = threadIdx.x >> 5, lane = threadIdx.x & 31;
  const int idx = blockIdx.x * 8 + wave;
  const int l = idx / H, h = idx % H;
  const float* src = raw + (size_t)l * inLd + h * inStride;
  float v[8]; float ss = 0.f;
#pragma unroll
  for (int i = 0; i < 8; ++i) { v[i] = src[lane + 32 * i]; ss += v[i] * v[i]; }
#pragma unroll
  for (int off = 1; off < 32; off <<= 1) ss += __shfl_xor(ss, off, 32);
  const float r = rsqrtf(ss * (1.0f / 256.0f) + 1e-6f);
  float nv[8];
#pragma unroll
  for (int i = 0; i < 8; ++i) nv[i] = v[i] * r * gamma[lane + 32 * i];
  const float c0 = cosT[l * ROPE_DIM + lane],      s0 = sinT[l * ROPE_DIM + lane];
  const float c1 = cosT[l * ROPE_DIM + lane + 32], s1 = sinT[l * ROPE_DIM + lane + 32];
  const float r0 = nv[0] * c0 - nv[1] * s0;
  const float r1 = nv[1] * c1 + nv[0] * s1;
  nv[0] = r0; nv[1] = r1;
  u16* dst = out + (size_t)l * outLd + h * HEAD_DIM;
#pragma unroll
  for (int i = 0; i < 8; ++i) dst[lane + 32 * i] = f2bf(nv[i]);
}

// ---------------------------------------------------------------------------
// Flash attention (causal, GQA) + sigmoid gate.
// One wave per (head h, 16-row query block). 32-col K/V blocks, online softmax.
// S = Q K^T via WMMA, P staged through LDS to re-form an A-fragment, O += P V.
// ---------------------------------------------------------------------------
__global__ void __launch_bounds__(256) flash_attn_kernel(const u16* __restrict__ Qb,   // L x 2048
                                                         const u16* __restrict__ Kb,   // L x 512
                                                         const u16* __restrict__ Vt,   // 512 x L
                                                         const float* __restrict__ qraw, // L x 4096 (gate)
                                                         u16* __restrict__ Ob) {       // L x 2048
  __shared__ __align__(16) u16 lds_p[8][16 * 32];
  const int wave = threadIdx.x >> 5;
  const int lane = threadIdx.x & 31;
  const int gid = blockIdx.x * 8 + wave;    // 0..2047
  const int h  = gid >> 8;                  // 256 q-blocks per head
  const int qb = gid & 255;
  const int g  = h >> 2;                    // GROUP_SIZE = 4
  const int q0 = qb * 16;

  // Preload Q fragments (16 x 256, 8 k-steps) — reused across all K blocks.
  bf16x16 qf[8];
#pragma unroll
  for (int kk = 0; kk < 8; ++kk)
    qf[kk] = load_frag(Qb + (size_t)q0 * ODIM + h * HEAD_DIM + kk * 32, ODIM);

  v8f o[16];
#pragma unroll
  for (int t = 0; t < 16; ++t) o[t] = {};
  float mrow[8], lrow[8];
#pragma unroll
  for (int r = 0; r < 8; ++r) { mrow[r] = -1e30f; lrow[r] = 0.f; }

  u16* lp = &lds_p[wave][0];
  const float scale = 0.0625f;              // 256^-0.5

  for (int j0 = 0; j0 < q0 + 16; j0 += 32) {     // wave-uniform causal bound
    // ---- S = Q K^T (two 16x16 tiles covering 32 K-columns) ----
    v8f s0 = {}, s1 = {};
#pragma unroll
    for (int kk = 0; kk < 8; ++kk) {
      bf16x16 b0 = load_frag(Kb + (size_t)j0 * N_KV + g * HEAD_DIM + kk * 32, N_KV);
      bf16x16 b1 = load_frag(Kb + ((size_t)j0 + 16) * N_KV + g * HEAD_DIM + kk * 32, N_KV);
      s0 = wmma_bf16(qf[kk], b0, s0);
      s1 = wmma_bf16(qf[kk], b1, s1);
    }
    // ---- mask + online softmax (row = VGPR r + 8*(lane>=16)) ----
    const int col0 = j0 + (lane & 15);
    const int col1 = col0 + 16;
    const int rbase = q0 + ((lane >> 4) << 3);
    float alpha[8];
#pragma unroll
    for (int r = 0; r < 8; ++r) {
      const int row = rbase + r;
      float e0 = (col0 <= row) ? s0[r] * scale : -1e30f;
      float e1 = (col1 <= row) ? s1[r] * scale : -1e30f;
      float mx = fmaxf(e0, e1);
#pragma unroll
      for (int off = 1; off < 16; off <<= 1) mx = fmaxf(mx, __shfl_xor(mx, off, 32));
      const float mnew = fmaxf(mrow[r], mx);
      const float a = __expf(mrow[r] - mnew);
      const float p0 = __expf(e0 - mnew);
      const float p1 = __expf(e1 - mnew);
      float rs = p0 + p1;
#pragma unroll
      for (int off = 1; off < 16; off <<= 1) rs += __shfl_xor(rs, off, 32);
      lrow[r] = lrow[r] * a + rs;
      mrow[r] = mnew;
      alpha[r] = a;
      s0[r] = p0; s1[r] = p1;               // reuse as P
    }
#pragma unroll
    for (int t = 0; t < 16; ++t)
#pragma unroll
      for (int r = 0; r < 8; ++r) o[t][r] *= alpha[r];

    // ---- stage P (16x32) through LDS to rebuild an A-fragment ----
#pragma unroll
    for (int r = 0; r < 8; ++r) {
      const int m = ((lane >> 4) << 3) + r;
      lp[m * 32 + (lane & 15)]      = f2bf(s0[r]);
      lp[m * 32 + (lane & 15) + 16] = f2bf(s1[r]);
    }
    asm volatile("s_wait_dscnt 0" ::: "memory");
    __builtin_amdgcn_wave_barrier();
    bf16x16 pf = load_frag(lp, 32);
    __builtin_amdgcn_wave_barrier();

    // ---- O += P V  (V columns via Vt rows, row-major B^T) ----
#pragma unroll
    for (int t = 0; t < 16; ++t) {
      bf16x16 vf = load_frag(Vt + (size_t)(g * HEAD_DIM + t * 16) * L_SEQ + j0, L_SEQ);
      o[t] = wmma_bf16(pf, vf, o[t]);
    }
  }

  // ---- epilogue: 1/l, sigmoid gate (gate = qraw[:, h*512+256+d]), bf16 out ----
#pragma unroll
  for (int t = 0; t < 16; ++t) {
#pragma unroll
    for (int r = 0; r < 8; ++r) {
      const int row = q0 + ((lane >> 4) << 3) + r;
      const int d   = t * 16 + (lane & 15);
      float val = o[t][r] / lrow[r];
      const float gt = qraw[(size_t)row * (2 * ODIM) + h * (2 * HEAD_DIM) + HEAD_DIM + d];
      val *= 1.f / (1.f + __expf(-gt));
      Ob[(size_t)row * ODIM + h * HEAD_DIM + d] = f2bf(val);
    }
  }
}

// ---------------------------------------------------------------------------
// Host launcher
// ---------------------------------------------------------------------------
extern "C" void kernel_launch(void* const* d_in, const int* in_sizes, int n_in,
                              void* d_out, int out_size, void* d_ws, size_t ws_size,
                              hipStream_t stream) {
  (void)in_sizes; (void)n_in; (void)out_size; (void)ws_size;
  const float* x    = (const float*)d_in[0];
  const float* cosT = (const float*)d_in[1];
  const float* sinT = (const float*)d_in[2];
  // d_in[3] = mask (causal, recomputed on device — unused)
  const float* wq = (const float*)d_in[4];
  const float* wk = (const float*)d_in[5];
  const float* wv = (const float*)d_in[6];
  const float* wo = (const float*)d_in[7];
  const float* qg = (const float*)d_in[8];
  const float* kg = (const float*)d_in[9];
  float* out = (float*)d_out;

  // Workspace layout (bytes)
  char* ws = (char*)d_ws;
  size_t off = 0;
  auto alloc = [&](size_t bytes) { void* p = ws + off; off += (bytes + 255) & ~(size_t)255; return p; };
  u16*   xb   = (u16*)  alloc((size_t)L_SEQ * IDIM * 2);        // 16 MB
  u16*   wqT  = (u16*)  alloc((size_t)(2 * ODIM) * IDIM * 2);   // 16 MB
  u16*   wkT  = (u16*)  alloc((size_t)N_KV * IDIM * 2);         //  2 MB
  u16*   wvT  = (u16*)  alloc((size_t)N_KV * IDIM * 2);         //  2 MB
  u16*   woT  = (u16*)  alloc((size_t)IDIM * ODIM * 2);         //  8 MB
  float* qraw = (float*)alloc((size_t)L_SEQ * (2 * ODIM) * 4);  // 64 MB
  float* kraw = (float*)alloc((size_t)L_SEQ * N_KV * 4);        //  8 MB
  float* vraw = (float*)alloc((size_t)L_SEQ * N_KV * 4);        //  8 MB
  u16*   Qb   = (u16*)  alloc((size_t)L_SEQ * ODIM * 2);        // 16 MB
  u16*   Kb   = (u16*)  alloc((size_t)L_SEQ * N_KV * 2);        //  4 MB
  u16*   Vt   = (u16*)  alloc((size_t)N_KV * L_SEQ * 2);        //  4 MB
  u16*   Ob   = (u16*)  alloc((size_t)L_SEQ * ODIM * 2);        // 16 MB

  dim3 blk(256);

  // 1) precision conversion / weight transposes
  cvt_bf16_kernel<<<(L_SEQ * IDIM) / 256, blk, 0, stream>>>(x, xb, L_SEQ * IDIM);
  transpose_w_kernel<<<dim3((2 * ODIM) / 32, IDIM / 8), blk, 0, stream>>>(wq, wqT, IDIM, 2 * ODIM);
  transpose_w_kernel<<<dim3(N_KV / 32, IDIM / 8), blk, 0, stream>>>(wk, wkT, IDIM, N_KV);
  transpose_w_kernel<<<dim3(N_KV / 32, IDIM / 8), blk, 0, stream>>>(wv, wvT, IDIM, N_KV);
  transpose_w_kernel<<<dim3(IDIM / 32, ODIM / 8), blk, 0, stream>>>(wo, woT, ODIM, IDIM);

  // 2) projections (bf16 WMMA + TDM LDS staging, fp32 accum)
  gemm_bf16_tdm_kernel<<<dim3(L_SEQ / 128, (2 * ODIM) / 64), blk, 0, stream>>>(xb, wqT, qraw, L_SEQ, 2 * ODIM, IDIM);
  gemm_bf16_tdm_kernel<<<dim3(L_SEQ / 128, N_KV / 64), blk, 0, stream>>>(xb, wkT, kraw, L_SEQ, N_KV, IDIM);
  gemm_bf16_tdm_kernel<<<dim3(L_SEQ / 128, N_KV / 64), blk, 0, stream>>>(xb, wvT, vraw, L_SEQ, N_KV, IDIM);

  // 3) QK RMSNorm + RoPE -> bf16 ; V -> bf16 transposed
  norm_rope_kernel<<<(L_SEQ * N_HEADS) / 8, blk, 0, stream>>>(qraw, qg, cosT, sinT, Qb,
                                                              N_HEADS, 2 * HEAD_DIM, 2 * ODIM, ODIM);
  norm_rope_kernel<<<(L_SEQ * N_GROUPS) / 8, blk, 0, stream>>>(kraw, kg, cosT, sinT, Kb,
                                                               N_GROUPS, HEAD_DIM, N_KV, N_KV);
  vtrans_kernel<<<(L_SEQ * N_KV) / 256, blk, 0, stream>>>(vraw, Vt);

  // 4) flash attention + gate
  flash_attn_kernel<<<(N_HEADS * (L_SEQ / 16)) / 8, blk, 0, stream>>>(Qb, Kb, Vt, qraw, Ob);

  // 5) output projection -> fp32 d_out
  gemm_bf16_tdm_kernel<<<dim3(L_SEQ / 128, IDIM / 64), blk, 0, stream>>>(Ob, woT, out, L_SEQ, IDIM, ODIM);
}
